// NNUE_27049704030261
// MI455X (gfx1250) — 8-bit matrix operand order repaired and checked by reading, compiled, not run on hardware
//
#include <hip/hip_runtime.h>
#include <hip/hip_bf16.h>

typedef __attribute__((ext_vector_type(16))) int   v16i;
typedef __attribute__((ext_vector_type(8)))  float v8f;
typedef __attribute__((ext_vector_type(4)))  float v4f;

#define B_ROWS 4096
#define H_FEAT 20480
#define D_FEAT (2 * H_FEAT)
#define N_OUT  512
#define KC8    256
#define LDB8   272

union AFrag { v16i v; unsigned long long u[8]; };
union BFrag { v16i v; uint4 q[4]; };

__device__ __forceinline__ unsigned char f32_to_e4m3(float x) {
    unsigned char s = (x < 0.0f) ? 0x80 : 0x00;
    float a = fabsf(x);
    if (!(a > 0.0f)) return s;
    if (a >= 448.0f) return s | 0x7E;
    int e; float f = frexpf(a, &e);
    (void)f;
    if (e >= -5) {
        int q = (int)rintf(ldexpf(a, 4 - e));
        if (q == 16) { q = 8; ++e; }
        int E = e + 6;
        if (E >= 15) return s | 0x7E;
        return s | (unsigned char)((E << 3) | (q & 7));
    }
    int q = (int)rintf(ldexpf(a, 9));
    return s | (unsigned char)q;
}

__global__ void __launch_bounds__(256)
nnue_convert_weights_fp8(const float* __restrict__ Ww,
                         const float* __restrict__ Wb,
                         unsigned char* __restrict__ W8) {
    size_t tid = (size_t)blockIdx.x * blockDim.x + threadIdx.x;
    size_t e   = tid * 8;
    int n = (int)(e / D_FEAT);
    int k = (int)(e % D_FEAT);
    const float* src;
    if (n < 256) {
        src = Ww + (size_t)n * D_FEAT + k;
    } else {
        int k2 = (k < H_FEAT) ? (k + H_FEAT) : (k - H_FEAT);
        src = Wb + (size_t)(n - 256) * D_FEAT + k2;
    }
    float4 f0 = ((const float4*)src)[0];
    float4 f1 = ((const float4*)src)[1];
    unsigned long long out = 0;
    out |= (unsigned long long)f32_to_e4m3(f0.x);
    out |= (unsigned long long)f32_to_e4m3(f0.y) << 8;
    out |= (unsigned long long)f32_to_e4m3(f0.z) << 16;
    out |= (unsigned long long)f32_to_e4m3(f0.w) << 24;
    out |= (unsigned long long)f32_to_e4m3(f1.x) << 32;
    out |= (unsigned long long)f32_to_e4m3(f1.y) << 40;
    out |= (unsigned long long)f32_to_e4m3(f1.z) << 48;
    out |= (unsigned long long)f32_to_e4m3(f1.w) << 56;
    *(unsigned long long*)(W8 + e) = out;
}

__global__ void __launch_bounds__(256)
nnue_ft_gemm_fp8(const float* __restrict__ white,
                 const float* __restrict__ black,
                 const unsigned char* __restrict__ W8,
                 const float* __restrict__ bw,
                 const float* __restrict__ bb,
                 float* __restrict__ out512) {
    __shared__ unsigned char aLds[32 * LDB8];

    const int tid    = threadIdx.x;
    const int lane   = tid & 31;
    const int wv     = tid >> 5;
    const int hi     = lane >> 4;
    const int ln     = lane & 15;
    const int blockM = blockIdx.x * 32;

    v8f acc[2][4];
#pragma unroll
    for (int mt = 0; mt < 2; ++mt)
#pragma unroll
        for (int nt = 0; nt < 4; ++nt)
#pragma unroll
            for (int r = 0; r < 8; ++r)
                acc[mt][nt][r] = 0.0f;

    const int    mL    = tid >> 3;
    const int    kloc  = (tid & 7) * 32;
    const size_t gmRow = (size_t)(blockM + mL) * H_FEAT;
    unsigned char* dst = &aLds[mL * LDB8 + kloc];

    #define ACT_PTR(kk) \
        (((kk) < H_FEAT) ? (white + gmRow + (kk)) : (black + gmRow + ((kk) - H_FEAT)))

    v4f fbuf[8];
    {
        const v4f* s0 = (const v4f*)ACT_PTR(kloc);
#pragma unroll
        for (int g = 0; g < 8; ++g)
            fbuf[g] = __builtin_nontemporal_load(s0 + g);
    }

    for (int kc = 0; kc < D_FEAT; kc += KC8) {
        {
            unsigned int w[8];
#pragma unroll
            for (int g = 0; g < 8; ++g) {
                v4f f = fbuf[g];
                w[g] = (f.x != 0.0f ? 0x00000038u : 0u)
                     | (f.y != 0.0f ? 0x00003800u : 0u)
                     | (f.z != 0.0f ? 0x00380000u : 0u)
                     | (f.w != 0.0f ? 0x38000000u : 0u);
            }
            ((uint4*)dst)[0] = make_uint4(w[0], w[1], w[2], w[3]);
            ((uint4*)dst)[1] = make_uint4(w[4], w[5], w[6], w[7]);
        }
        __syncthreads();

        const int  knext = kc + KC8;
        const bool more  = knext < D_FEAT;

#pragma unroll
        for (int ks = 0; ks < KC8; ks += 128) {
            AFrag af[2];
#pragma unroll
            for (int mt = 0; mt < 2; ++mt) {
                const unsigned char* ap = &aLds[(mt * 16 + ln) * LDB8 + ks + hi * 16];
#pragma unroll
                for (int p = 0; p < 8; ++p) {
                    int off = (p >> 1) * 32 + (p & 1) * 8;
                    af[mt].u[p] = *(const unsigned long long*)(ap + off);
                }
            }
#pragma unroll
            for (int nt = 0; nt < 4; ++nt) {
                int n = wv * 64 + nt * 16 + ln;
                const unsigned char* bp =
                    W8 + (size_t)n * D_FEAT + (kc + ks) + hi * 16;
                BFrag bf;
#pragma unroll
                for (int j = 0; j < 4; ++j)
                    bf.q[j] = *(const uint4*)(bp + j * 32);
                acc[0][nt] = __builtin_amdgcn_wmma_f32_16x16x128_fp8_fp8(
                    af[0].v, bf.v, (short)0, acc[0][nt], false, false);
                acc[1][nt] = __builtin_amdgcn_wmma_f32_16x16x128_fp8_fp8(
                    af[1].v, bf.v, (short)0, acc[1][nt], false, false);
            }

            if (ks == 0 && more) {
                const v4f* sn = (const v4f*)ACT_PTR(knext + kloc);
#pragma unroll
                for (int g = 0; g < 8; ++g)
                    fbuf[g] = __builtin_nontemporal_load(sn + g);
                int k2 = knext + KC8;
                if (k2 < D_FEAT)
                    __builtin_prefetch(ACT_PTR(k2 + kloc), 0, 0);
            }
        }
        __syncthreads();
    }
    #undef ACT_PTR

#pragma unroll
    for (int nt = 0; nt < 4; ++nt) {
        int n = wv * 64 + nt * 16 + ln;
        float biasv = (n < 256) ? bw[n] : bb[n - 256];
#pragma unroll
        for (int mt = 0; mt < 2; ++mt) {
            int gm0 = blockM + mt * 16 + hi * 8;
#pragma unroll
            for (int r = 0; r < 8; ++r) {
                out512[(size_t)(gm0 + r) * N_OUT + n] = acc[mt][nt][r] + biasv;
            }
        }
    }
}

__global__ void __launch_bounds__(256)
nnue_tail(const float* __restrict__ out512,
          const float* __restrict__ pov,
          const float* __restrict__ W0, const float* __restrict__ b0,
          const float* __restrict__ W1, const float* __restrict__ b1,
          const float* __restrict__ W2, const float* __restrict__ b2,
          float* __restrict__ y) {
    __shared__ float xs[8][512];
    const int lane = threadIdx.x & 31;
    const int wv   = threadIdx.x >> 5;
    const int m    = blockIdx.x * 8 + wv;

    const float* row = out512 + (size_t)m * N_OUT;
    int rot = (pov[m] >= 0.5f) ? 0 : 256;
#pragma unroll
    for (int j = lane; j < 512; j += 32) {
        float v = row[(j + rot) & 511];
        xs[wv][j] = fmaxf(v, 0.0f);
    }
    __syncthreads();

    float a0 = b0[lane];
    const float* w0r = W0 + (size_t)lane * 512;
    for (int j = 0; j < 512; ++j)
        a0 = fmaf(w0r[j], xs[wv][j], a0);
    float h0 = fmaxf(a0, 0.0f);

    float a1 = b1[lane];
    const float* w1r = W1 + (size_t)lane * 32;
#pragma unroll
    for (int j = 0; j < 32; ++j)
        a1 = fmaf(w1r[j], __shfl(h0, j, 32), a1);
    float h1 = fmaxf(a1, 0.0f);

    float part = W2[lane] * h1;
#pragma unroll
    for (int off = 16; off > 0; off >>= 1)
        part += __shfl_down(part, off, 32);
    if (lane == 0)
        y[m] = part + b2[0];
}

extern "C" void kernel_launch(void* const* d_in, const int* in_sizes, int n_in,
                              void* d_out, int out_size, void* d_ws, size_t ws_size,
                              hipStream_t stream) {
    const float* pov   = (const float*)d_in[0];
    const float* white = (const float*)d_in[1];
    const float* black = (const float*)d_in[2];
    const float* Ww    = (const float*)d_in[3];
    const float* bw    = (const float*)d_in[4];
    const float* Wb    = (const float*)d_in[5];
    const float* bb    = (const float*)d_in[6];
    const float* W0    = (const float*)d_in[7];
    const float* b0    = (const float*)d_in[8];
    const float* W1    = (const float*)d_in[9];
    const float* b1    = (const float*)d_in[10];
    const float* W2    = (const float*)d_in[11];
    const float* b2    = (const float*)d_in[12];
    float* y = (float*)d_out;

    char* ws = (char*)d_ws;
    unsigned char* W8   = (unsigned char*)ws;
    float*         o512 = (float*)(ws + (size_t)N_OUT * D_FEAT);

    nnue_convert_weights_fp8<<<dim3(10240), dim3(256), 0, stream>>>(Ww, Wb, W8);

    nnue_ft_gemm_fp8<<<dim3(B_ROWS / 32), dim3(256), 0, stream>>>(
        white, black, W8, bw, bb, o512);

    nnue_tail<<<dim3(B_ROWS / 8), dim3(256), 0, stream>>>(
        o512, pov, W0, b0, W1, b1, W2, b2, y);
}
